// CausalSelfAttention_12335146074412
// MI455X (gfx1250) — matrix operand order repaired and checked by reading, not compiled, not run on hardware
//
#include <hip/hip_runtime.h>
#include <hip/hip_bf16.h>

// ---------------------------------------------------------------------------
// Types for CDNA5 WMMA (wave32): v_wmma_f32_16x16x32_bf16
// ---------------------------------------------------------------------------
typedef __attribute__((ext_vector_type(16))) __bf16 v16bf;
typedef __attribute__((ext_vector_type(8)))  float  v8f;

static __device__ __forceinline__ v8f vzero8() {
    v8f v = {0.f, 0.f, 0.f, 0.f, 0.f, 0.f, 0.f, 0.f};
    return v;
}

static __device__ __forceinline__ v8f wmma_bf16(v16bf a, v16bf b, v8f c) {
    // (neg_a, A, neg_b, B, c_mod, C, reuse_a, reuse_b)
    return __builtin_amdgcn_wmma_f32_16x16x32_bf16(false, a, false, b, (short)0, c, false, false);
}

// fp32 -> bf16 (round to nearest even), kept as raw ushort for storage
static __device__ __forceinline__ unsigned short f2bf(float f) {
    unsigned u = __float_as_uint(f);
    u += 0x7FFFu + ((u >> 16) & 1u);
    return (unsigned short)(u >> 16);
}

union FragU {
    v16bf v;
    uint4 q[2];
};

// Assemble a 16-element bf16 fragment from two 16-byte chunks (LDS or global)
static __device__ __forceinline__ v16bf load_frag(const unsigned short* p0,
                                                  const unsigned short* p1) {
    FragU f;
    f.q[0] = *(const uint4*)p0;
    f.q[1] = *(const uint4*)p1;
    return f.v;
}

// ---------------------------------------------------------------------------
// CDNA5 async global->LDS (DMA path, tracked by ASYNCcnt; bypasses VGPRs)
// ---------------------------------------------------------------------------
static __device__ __forceinline__ void async_load_b128(void* lds_ptr,
                                                       const void* gaddr) {
    unsigned lds_off = (unsigned)(unsigned long long)(uintptr_t)lds_ptr;  // low 32 = LDS byte offset
    unsigned long long ga = (unsigned long long)(uintptr_t)gaddr;
    asm volatile("global_load_async_to_lds_b128 %0, %1, off"
                 :
                 : "v"(lds_off), "v"(ga)
                 : "memory");
}

static __device__ __forceinline__ void wait_async0() {
    asm volatile("s_wait_asynccnt 0" ::: "memory");
}

// ---------------------------------------------------------------------------
// 16-lane butterfly reductions via ds_swizzle (xor_mask in offset[14:10])
// ---------------------------------------------------------------------------
static __device__ __forceinline__ float red_max16(float v) {
    v = fmaxf(v, __int_as_float(__builtin_amdgcn_ds_swizzle(__float_as_int(v), 0x041F)));
    v = fmaxf(v, __int_as_float(__builtin_amdgcn_ds_swizzle(__float_as_int(v), 0x081F)));
    v = fmaxf(v, __int_as_float(__builtin_amdgcn_ds_swizzle(__float_as_int(v), 0x101F)));
    v = fmaxf(v, __int_as_float(__builtin_amdgcn_ds_swizzle(__float_as_int(v), 0x201F)));
    return v;
}
static __device__ __forceinline__ float red_sum16(float v) {
    v += __int_as_float(__builtin_amdgcn_ds_swizzle(__float_as_int(v), 0x041F));
    v += __int_as_float(__builtin_amdgcn_ds_swizzle(__float_as_int(v), 0x081F));
    v += __int_as_float(__builtin_amdgcn_ds_swizzle(__float_as_int(v), 0x101F));
    v += __int_as_float(__builtin_amdgcn_ds_swizzle(__float_as_int(v), 0x201F));
    return v;
}

// ---------------------------------------------------------------------------
// Prep kernels: fp32 -> bf16 convert, and transposed convert for weights
// ---------------------------------------------------------------------------
__global__ void convert_f32_bf16(const float* __restrict__ in,
                                 unsigned short* __restrict__ out, int n) {
    int i = blockIdx.x * blockDim.x + threadIdx.x;
    if (i < n) out[i] = f2bf(in[i]);
}

// in [K][N] fp32 -> out [N][K] bf16  (so GEMM B-fragments are contiguous in K)
__global__ void transpose_f32_bf16(const float* __restrict__ in,
                                   unsigned short* __restrict__ out,
                                   int K, int N) {
    int i = blockIdx.x * blockDim.x + threadIdx.x;
    if (i < K * N) {
        int n = i % N;
        int k = i / N;
        out[(size_t)n * K + k] = f2bf(in[i]);
    }
}

// ---------------------------------------------------------------------------
// bf16 WMMA GEMM:  C[M][N] = A[M][K] * Bt[N][K]^T + bias[N]
// Block tile 128x128, BK=32. 256 threads = 8 waves; each wave 32(M) x 64(N).
// Double-buffered LDS fed by async global->LDS DMA.
// ---------------------------------------------------------------------------
template <bool OUT_BF16>
__global__ __launch_bounds__(256) void gemm_bf16_wmma(
    const unsigned short* __restrict__ A,   // [M][K] bf16
    const unsigned short* __restrict__ Bt,  // [N][K] bf16 (pre-transposed)
    const float* __restrict__ bias,         // [N]
    void* __restrict__ Cout, int M, int N, int K) {
    constexpr int BM = 128, BN = 128, BK = 32;
    __shared__ __align__(16) unsigned short sA[2][BM * BK];  // 2 x 8 KB
    __shared__ __align__(16) unsigned short sB[2][BN * BK];  // 2 x 8 KB

    const int tid  = threadIdx.x;
    const int lane = tid & 31;
    const int wave = tid >> 5;  // 0..7
    const int m0 = (int)blockIdx.y * BM;
    const int n0 = (int)blockIdx.x * BN;
    const int wm = (wave & 3) * 32;   // wave M offset in tile
    const int wn = (wave >> 2) * 64;  // wave N offset in tile

    v8f acc[2][4];
#pragma unroll
    for (int i = 0; i < 2; ++i)
#pragma unroll
        for (int j = 0; j < 4; ++j) acc[i][j] = vzero8();

    // per-lane fragment addressing (ISA 7.12.2 layouts, wave32)
    const int a_m  = lane & 15;          // A: lane -> row
    const int a_kb = (lane >> 4) * 8;    // A: K base {0,8}; elems +0..7 and +16..23
    const int b_n  = lane & 15;          // B: lane -> column
    const int b_kb = (lane >> 4) * 8;    // B: K base {0,8}; elems +0..7 and +16..23

    const int lrow = tid >> 1;           // 0..127 tile row for staging
    const int lcol = (tid & 1) * 16;     // 0 or 16 within BK

    const unsigned short* gArow = A  + (size_t)(m0 + lrow) * K + lcol;
    const unsigned short* gBrow = Bt + (size_t)(n0 + lrow) * K + lcol;

    auto stage = [&](int buf, int k0) {
        async_load_b128(&sA[buf][lrow * BK + lcol],     gArow + k0);
        async_load_b128(&sA[buf][lrow * BK + lcol + 8], gArow + k0 + 8);
        async_load_b128(&sB[buf][lrow * BK + lcol],     gBrow + k0);
        async_load_b128(&sB[buf][lrow * BK + lcol + 8], gBrow + k0 + 8);
    };

    const int KT = K / BK;
    stage(0, 0);
    wait_async0();
    __syncthreads();

    for (int kt = 0; kt < KT; ++kt) {
        const int cur = kt & 1;
        if (kt + 1 < KT) stage((kt + 1) & 1, (kt + 1) * BK);   // DMA next tile
        if (kt + 2 < KT) {                                     // warm L2 for tile kt+2
            __builtin_prefetch(gArow + (kt + 2) * BK, 0, 1);
            __builtin_prefetch(gBrow + (kt + 2) * BK, 0, 1);
        }

        v16bf bf[4];
#pragma unroll
        for (int j = 0; j < 4; ++j) {
            const unsigned short* p = &sB[cur][(wn + j * 16 + b_n) * BK + b_kb];
            bf[j] = load_frag(p, p + 16);
        }
#pragma unroll
        for (int i = 0; i < 2; ++i) {
            const unsigned short* pA = &sA[cur][(wm + i * 16 + a_m) * BK + a_kb];
            v16bf af = load_frag(pA, pA + 16);
#pragma unroll
            for (int j = 0; j < 4; ++j) acc[i][j] = wmma_bf16(af, bf[j], acc[i][j]);
        }

        if (kt + 1 < KT) wait_async0();  // our DMA into the other buffer done
        __syncthreads();                 // everyone done reading cur + DMA visible
    }

    // epilogue: C layout — VGPR r: lanes 0-15 row r, lanes 16-31 row r+8
    const int c_n  = lane & 15;
    const int mofs = (lane >> 4) * 8;
#pragma unroll
    for (int i = 0; i < 2; ++i) {
#pragma unroll
        for (int j = 0; j < 4; ++j) {
            const int col = n0 + wn + j * 16 + c_n;
            const float bv = bias[col];
            const int rowbase = m0 + wm + i * 16 + mofs;
#pragma unroll
            for (int r = 0; r < 8; ++r) {
                float v = acc[i][j][r] + bv;
                size_t idx = (size_t)(rowbase + r) * N + col;
                if (OUT_BF16)
                    ((unsigned short*)Cout)[idx] = f2bf(v);
                else
                    ((float*)Cout)[idx] = v;
            }
        }
    }
}

// ---------------------------------------------------------------------------
// Causal flash attention (bf16 WMMA, fp32 online softmax).
// Grid: x = L/64 q-blocks, y = B*H. Block = 128 threads (4 waves x 16 q-rows).
// qkv: [B, L, 3*C] bf16 with c = part*C + h*64 + dh. y: [B, L, C] bf16.
// ---------------------------------------------------------------------------
__global__ __launch_bounds__(128) void attn_causal_wmma(
    const unsigned short* __restrict__ qkv, unsigned short* __restrict__ y) {
    constexpr int L = 2048, Cm = 1024, Dh = 64, C3 = 3072;
    __shared__ __align__(16) unsigned short sK[32 * 64];      // [key][dh]   4 KB
    __shared__ __align__(16) unsigned short sVt[64 * 32];     // [dh][key]   4 KB
    __shared__ __align__(16) unsigned short sP[4][16 * 32];   // per-wave P  4 KB

    const int tid  = threadIdx.x;
    const int lane = tid & 31;
    const int wave = tid >> 5;  // 0..3
    const int bh = blockIdx.y;
    const int b = bh >> 4, h = bh & 15;
    const int q0 = (int)blockIdx.x * 64;
    const int qw = q0 + wave * 16;  // this wave's first q row

    const int a_m  = lane & 15;
    const int a_kb = (lane >> 4) * 8;
    const int c_n  = lane & 15;
    const int mofs = (lane >> 4) * 8;
    const int b_kb = (lane >> 4) * 8;

    // Q fragments (A-layout), kept in registers: 16 rows x 64 dh -> two K=32 steps
    v16bf qf[2];
    {
        const unsigned short* qrow =
            qkv + (size_t)(b * L + qw + a_m) * C3 + h * Dh;
#pragma unroll
        for (int s = 0; s < 2; ++s) {
            const unsigned short* p = qrow + s * 32 + a_kb;
            qf[s] = load_frag(p, p + 16);
        }
    }

    float mrow[8], lrow[8];
    v8f o[4];
#pragma unroll
    for (int r = 0; r < 8; ++r) { mrow[r] = -1e30f; lrow[r] = 0.f; }
#pragma unroll
    for (int t = 0; t < 4; ++t) o[t] = vzero8();

    const unsigned short* Kbase = qkv + (size_t)(b * L) * C3 + Cm + h * Dh;
    const unsigned short* Vbase = qkv + (size_t)(b * L) * C3 + 2 * Cm + h * Dh;

    const int qmax_wave = qw + 15;
    const int nblocks = (q0 + 64) >> 5;  // 32-key blocks covering keys <= q0+63

    const int skey = tid >> 2;           // staging: key row 0..31
    const int sdh  = (tid & 3) * 16;     // staging: dh chunk

    for (int jb = 0; jb < nblocks; ++jb) {
        const int j = jb * 32;
        // K tile via async DMA (global -> LDS, no VGPR round trip)
        {
            const unsigned short* gk = Kbase + (size_t)(j + skey) * C3 + sdh;
            async_load_b128(&sK[skey * 64 + sdh],     gk);
            async_load_b128(&sK[skey * 64 + sdh + 8], gk + 8);
        }
        // V tile, transposed into LDS [dh][key] (register unpack, 16 b16 stores)
        {
            const uint4* gv = (const uint4*)(Vbase + (size_t)(j + skey) * C3 + sdh);
            union { uint4 q[2]; unsigned short u[16]; } vv;
            vv.q[0] = gv[0];
            vv.q[1] = gv[1];
#pragma unroll
            for (int e = 0; e < 16; ++e) sVt[(sdh + e) * 32 + skey] = vv.u[e];
        }
        wait_async0();
        __syncthreads();

        if (j <= qmax_wave) {
            // S = Q @ K^T : two 16x16 score tiles (keys j..j+15, j+16..j+31)
            v8f s0 = vzero8(), s1 = vzero8();
#pragma unroll
            for (int s = 0; s < 2; ++s) {
                const unsigned short* pb0 = &sK[c_n * 64 + s * 32 + b_kb];
                s0 = wmma_bf16(qf[s], load_frag(pb0, pb0 + 16), s0);
                const unsigned short* pb1 = &sK[(16 + c_n) * 64 + s * 32 + b_kb];
                s1 = wmma_bf16(qf[s], load_frag(pb1, pb1 + 16), s1);
            }

            const float scale = 0.125f;  // 1/sqrt(64)
            const int k0i = j + c_n;
            const int k1i = j + 16 + c_n;
            float rmax[8];
#pragma unroll
            for (int r = 0; r < 8; ++r) {
                const int qi = qw + mofs + r;
                float v0 = s0[r] * scale;
                float v1 = s1[r] * scale;
                if (k0i > qi) v0 = -1e30f;
                if (k1i > qi) v1 = -1e30f;
                s0[r] = v0; s1[r] = v1;
                rmax[r] = fmaxf(v0, v1);
            }

            unsigned short* P = &sP[wave][0];
#pragma unroll
            for (int r = 0; r < 8; ++r) {
                const float mx    = red_max16(rmax[r]);
                const float mnew  = fmaxf(mrow[r], mx);
                const float alpha = __expf(mrow[r] - mnew);
                mrow[r] = mnew;
                const float p0 = __expf(s0[r] - mnew);
                const float p1 = __expf(s1[r] - mnew);
                const float rs = red_sum16(p0 + p1);
                lrow[r] = lrow[r] * alpha + rs;
                o[0][r] *= alpha; o[1][r] *= alpha;
                o[2][r] *= alpha; o[3][r] *= alpha;
                // write P (C layout -> LDS row-major [m][key])
                const int m = mofs + r;
                P[m * 32 + c_n]      = f2bf(p0);
                P[m * 32 + 16 + c_n] = f2bf(p1);
            }

            // re-read P as A-fragment (16x32), then O += P @ V  (K=32)
            const unsigned short* pp = &P[a_m * 32 + a_kb];
            v16bf pf = load_frag(pp, pp + 16);
#pragma unroll
            for (int t = 0; t < 4; ++t) {
                const unsigned short* pv = &sVt[(t * 16 + c_n) * 32 + b_kb];
                o[t] = wmma_bf16(pf, load_frag(pv, pv + 16), o[t]);
            }
        }
        __syncthreads();
    }

    // epilogue: O /= l, store bf16 into y[b, q, h*64 + dh]
#pragma unroll
    for (int r = 0; r < 8; ++r) {
        const float inv = 1.0f / lrow[r];
        const int q = qw + mofs + r;
        unsigned short* yr = y + (size_t)(b * L + q) * Cm + h * Dh;
#pragma unroll
        for (int t = 0; t < 4; ++t) yr[t * 16 + c_n] = f2bf(o[t][r] * inv);
    }
}

// ---------------------------------------------------------------------------
// Host-side launcher
// ---------------------------------------------------------------------------
extern "C" void kernel_launch(void* const* d_in, const int* in_sizes, int n_in,
                              void* d_out, int out_size, void* d_ws, size_t ws_size,
                              hipStream_t stream) {
    (void)in_sizes; (void)n_in; (void)out_size; (void)ws_size;
    const float* x     = (const float*)d_in[0];  // [2, 2048, 1024]
    const float* Wqkv  = (const float*)d_in[1];  // [1024, 3072]
    const float* bqkv  = (const float*)d_in[2];  // [3072]
    const float* Wproj = (const float*)d_in[3];  // [1024, 1024]
    const float* bproj = (const float*)d_in[4];  // [1024]
    float* out = (float*)d_out;

    constexpr int B = 2, L = 2048, Cm = 1024, C3 = 3072;
    constexpr int M = B * L;  // 4096

    char* ws = (char*)d_ws;
    unsigned short* xb   = (unsigned short*)(ws);                         //  8 MB
    unsigned short* Wqt  = (unsigned short*)(ws + (size_t)(8u  << 20));   //  6 MB
    unsigned short* Wpt  = (unsigned short*)(ws + (size_t)(14u << 20));   //  2 MB
    unsigned short* qkvb = (unsigned short*)(ws + (size_t)(16u << 20));   // 24 MB
    unsigned short* yb   = (unsigned short*)(ws + (size_t)(40u << 20));   //  8 MB

    // fp32 -> bf16 prep
    convert_f32_bf16<<<(M * Cm + 255) / 256, 256, 0, stream>>>(x, xb, M * Cm);
    transpose_f32_bf16<<<(Cm * C3 + 255) / 256, 256, 0, stream>>>(Wqkv, Wqt, Cm, C3);
    transpose_f32_bf16<<<(Cm * Cm + 255) / 256, 256, 0, stream>>>(Wproj, Wpt, Cm, Cm);

    // qkv = x @ W_qkv + b_qkv   (bf16 out)
    dim3 g1(C3 / 128, M / 128);
    gemm_bf16_wmma<true><<<g1, 256, 0, stream>>>(xb, Wqt, bqkv, qkvb, M, C3, Cm);

    // causal attention -> y (bf16)
    dim3 ga(L / 64, B * 16);
    attn_causal_wmma<<<ga, 128, 0, stream>>>(qkvb, yb);

    // out = y @ W_proj + b_proj (fp32 out)
    dim3 g2(Cm / 128, M / 128);
    gemm_bf16_wmma<false><<<g2, 256, 0, stream>>>(yb, Wpt, bproj, out, M, Cm, Cm);
}
